// MultiHeadAttention_41858751267402
// MI455X (gfx1250) — matrix operand order repaired and checked by reading, not compiled, not run on hardware
//
#include <hip/hip_runtime.h>
#include <stdint.h>

#define BATCH 4
#define S_LEN 1024
#define DM    1024
#define NH    16
#define DK    64

// padded LDS strides (elements) to avoid bank conflicts on 64x4B banks
#define AS_STRIDE 40    // A tile rows (80B)
#define BS_STRIDE 72    // B tile rows (144B)
#define SD_STRIDE 72    // D staging rows (144B)
#define SQ_STRIDE 72    // Q tile rows (144B)
#define SKT_STRIDE 136  // K^T rows (272B)
#define SP_STRIDE 1028  // score rows in f32 (4112B)

typedef __attribute__((ext_vector_type(16))) __bf16 v16bf;
typedef __attribute__((ext_vector_type(2)))  __bf16 v2bf;
typedef __attribute__((ext_vector_type(2)))  float  v2f;
typedef __attribute__((ext_vector_type(8)))  float  v8f;

union Frag {
  v16bf bf;
  uint4 u4[2];
};

// native f32->bf16 (RNE) — lowers to v_cvt_pk_bf16_f32-class ops
__device__ __forceinline__ unsigned short f2bf(float x) {
  union { __bf16 b; unsigned short u; } cv;
  cv.b = (__bf16)x;
  return cv.u;
}

__device__ __forceinline__ unsigned int pk2(float a, float b) {
  union { v2bf b; unsigned int u; } cv;
  v2f f = {a, b};
  cv.b = __builtin_convertvector(f, v2bf);
  return cv.u;
}

__device__ __forceinline__ v8f wmma_bf16(Frag a, Frag b, v8f c) {
  return __builtin_amdgcn_wmma_f32_16x16x32_bf16(
      /*neg_a=*/false, a.bf, /*neg_b=*/false, b.bf,
      /*c_mod=*/(short)0, c, /*reuse_a=*/false, /*reuse_b=*/false);
}

// ---------------------------------------------------------------------------
// Kernel 1: Q/K/V projection GEMM, f32 in -> bf16 out laid out [B,H,S,64].
// grid = (N/64, M/128, 3), block = 256 (8 waves, wave tile 32x32).
// Q is pre-scaled by 1/sqrt(DK)=0.125 (exact in bf16).
// Epilogue stages the 128x64 bf16 tile in LDS, then stores coalesced b128.
// ---------------------------------------------------------------------------
__global__ __launch_bounds__(256)
void qkv_proj_kernel(const float* __restrict__ query,
                     const float* __restrict__ context,
                     const float* __restrict__ Wq,
                     const float* __restrict__ Wk,
                     const float* __restrict__ Wv,
                     unsigned short* __restrict__ Qs,
                     unsigned short* __restrict__ Ks,
                     unsigned short* __restrict__ Vs)
{
  const int which = blockIdx.z;
  const float* __restrict__ A = (which == 1) ? context : query;
  const float* __restrict__ W = (which == 0) ? Wq : ((which == 1) ? Wk : Wv);
  unsigned short* __restrict__ Out = (which == 0) ? Qs : ((which == 1) ? Ks : Vs);
  const float scaleA = (which == 0) ? 0.125f : 1.0f;

  __shared__ union {
    struct {
      unsigned short As[128 * AS_STRIDE];  // [m][k] bf16
      unsigned short Bs[32 * BS_STRIDE];   // [k][n] bf16
    } t;
    unsigned short D[128 * SD_STRIDE];     // output staging (reuses tile space)
  } sm;

  const int tid  = threadIdx.x;
  const int lane = tid & 31;
  const int wave = tid >> 5;
  const int wm   = wave & 3;      // 0..3 -> 32-row strip
  const int wn   = wave >> 2;     // 0..1 -> 32-col strip
  const int m0   = blockIdx.y * 128;
  const int n0   = blockIdx.x * 64;

  const v8f vzero = {0.f, 0.f, 0.f, 0.f, 0.f, 0.f, 0.f, 0.f};
  v8f acc[2][2] = {{vzero, vzero}, {vzero, vzero}};

  const int arow  = tid >> 1;
  const int acb   = (tid & 1) * 16;
  const int brow  = tid >> 3;
  const int bcb   = (tid & 7) * 8;
  const int ahalf = (lane >> 4) * 8;   // K-half offset for A fragments
  const int al    = lane & 15;

  for (int kk = 0; kk < DM; kk += 32) {
    __syncthreads();
    {  // A tile 128x32: each thread converts 16 f32 -> bf16
      const float* src = A + (size_t)(m0 + arow) * DM + kk + acb;
      float4 v0 = *(const float4*)(src + 0);
      float4 v1 = *(const float4*)(src + 4);
      float4 v2 = *(const float4*)(src + 8);
      float4 v3 = *(const float4*)(src + 12);
      unsigned int* dst = (unsigned int*)&sm.t.As[arow * AS_STRIDE + acb];
      dst[0] = pk2(v0.x * scaleA, v0.y * scaleA);
      dst[1] = pk2(v0.z * scaleA, v0.w * scaleA);
      dst[2] = pk2(v1.x * scaleA, v1.y * scaleA);
      dst[3] = pk2(v1.z * scaleA, v1.w * scaleA);
      dst[4] = pk2(v2.x * scaleA, v2.y * scaleA);
      dst[5] = pk2(v2.z * scaleA, v2.w * scaleA);
      dst[6] = pk2(v3.x * scaleA, v3.y * scaleA);
      dst[7] = pk2(v3.z * scaleA, v3.w * scaleA);
    }
    {  // B tile 32x64: each thread converts 8 f32 -> bf16
      const float* src = W + (size_t)(kk + brow) * DM + n0 + bcb;
      float4 v0 = *(const float4*)(src + 0);
      float4 v1 = *(const float4*)(src + 4);
      unsigned int* dst = (unsigned int*)&sm.t.Bs[brow * BS_STRIDE + bcb];
      dst[0] = pk2(v0.x, v0.y);
      dst[1] = pk2(v0.z, v0.w);
      dst[2] = pk2(v1.x, v1.y);
      dst[3] = pk2(v1.z, v1.w);
    }
    __syncthreads();

    Frag af[2], bfg[2];
#pragma unroll
    for (int i = 0; i < 2; ++i) {
      const unsigned short* p = &sm.t.As[(wm * 32 + i * 16 + al) * AS_STRIDE + ahalf];
      af[i].u4[0] = *(const uint4*)(p);        // K = half..half+7
      af[i].u4[1] = *(const uint4*)(p + 16);   // K = half+16..half+23
    }
#pragma unroll
    for (int j = 0; j < 2; ++j) {
      const unsigned short* p = &sm.t.Bs[wn * 32 + j * 16 + al];  // N col = lane%16, K in table order
#pragma unroll
      for (int s = 0; s < 16; ++s) {
        int k = (s < 8) ? (ahalf + s) : (16 + ahalf + (s - 8));
        bfg[j].bf[s] = __builtin_bit_cast(__bf16, p[k * BS_STRIDE]);
      }
    }
#pragma unroll
    for (int i = 0; i < 2; ++i)
#pragma unroll
      for (int j = 0; j < 2; ++j)
        acc[i][j] = wmma_bf16(af[i], bfg[j], acc[i][j]);
  }

  // Epilogue: convert to bf16, stage tile in LDS, store coalesced 64B chunks.
  // D layout: vgpr r, lanes 0-15 -> (M=r, N=lane); lanes 16-31 -> (M=8+r, N=lane-16)
  const int hi = lane >> 4;
  const int nl = lane & 15;
  __syncthreads();   // As/Bs reads complete before overlaying D
#pragma unroll
  for (int i = 0; i < 2; ++i)
#pragma unroll
    for (int j = 0; j < 2; ++j)
#pragma unroll
      for (int r = 0; r < 8; ++r) {
        int m = wm * 32 + i * 16 + r + 8 * hi;
        int n = wn * 32 + j * 16 + nl;
        sm.D[m * SD_STRIDE + n] = f2bf(acc[i][j][r]);
      }
  __syncthreads();
  {
    const int row  = tid >> 1;          // local m (0..127)
    const int half = (tid & 1) * 32;    // d offset {0,32}
    const int M  = m0 + row;
    const int bb = M >> 10, ss = M & 1023;
    const int hh = n0 >> 6;             // one head per 64-wide N tile
    unsigned short* dstp =
        &Out[(((size_t)(bb * NH + hh)) * S_LEN + ss) * DK + half];
    const unsigned short* srcp = &sm.D[row * SD_STRIDE + half];
    *(uint4*)(dstp + 0)  = *(const uint4*)(srcp + 0);
    *(uint4*)(dstp + 8)  = *(const uint4*)(srcp + 8);
    *(uint4*)(dstp + 16) = *(const uint4*)(srcp + 16);
    *(uint4*)(dstp + 24) = *(const uint4*)(srcp + 24);
  }
}

// ---------------------------------------------------------------------------
// Kernel 2: attention for one (b, h, 32-row q tile).
// scores (WMMA) -> f32 LDS -> masked softmax -> write attn once ->
// bf16 probs in-place in LDS -> ctx = P @ V (WMMA) -> ctx f32 [B,S,D].
// Dynamic LDS: padded f32 scores (aliased by bf16 probs) + Q + K^T = 150KB.
// ---------------------------------------------------------------------------
__global__ __launch_bounds__(256)
void attn_kernel(const unsigned short* __restrict__ Qs,
                 const unsigned short* __restrict__ Ks,
                 const unsigned short* __restrict__ Vs,
                 const unsigned char*  __restrict__ mask,
                 float* __restrict__ attn_out,
                 float* __restrict__ ctx)
{
  extern __shared__ char smem[];
  float*          sP  = (float*)smem;                   // [32][SP_STRIDE] f32
  unsigned short* sPu = (unsigned short*)smem;          // alias, row stride 2*SP_STRIDE
  unsigned short* sQ  = (unsigned short*)(smem + 32 * SP_STRIDE * 4);  // [32][SQ_STRIDE]
  unsigned short* sKt = sQ + 32 * SQ_STRIDE;                           // [64][SKT_STRIDE]

  const int tid  = threadIdx.x;
  const int lane = tid & 31;
  const int wave = tid >> 5;
  const int q0   = blockIdx.x * 32;
  const int h    = blockIdx.y;
  const int b    = blockIdx.z;
  const size_t headBase = ((size_t)(b * NH + h)) * S_LEN * DK;

  {  // load Q tile (already scaled by 1/8)
    int row = tid >> 3, db = (tid & 7) * 8;
    *(uint4*)&sQ[row * SQ_STRIDE + db] =
        *(const uint4*)&Qs[headBase + (size_t)(q0 + row) * DK + db];
  }

  // warm the mask rows this wave will reduce later (global_prefetch_b8)
#pragma unroll
  for (int i = 0; i < 4; ++i) {
    const unsigned char* mrow =
        mask + ((size_t)b * S_LEN + q0 + wave * 4 + i) * S_LEN;
    __builtin_prefetch(mrow + lane * 32, 0, 0);
  }

  const int ahalf = (lane >> 4) * 8;
  const int al    = lane & 15;
  const int hi    = lane >> 4;
  const v8f vzero = {0.f, 0.f, 0.f, 0.f, 0.f, 0.f, 0.f, 0.f};

  // ---- scores: S = Q K^T over 8 chunks of 128 key columns ----
  for (int nc = 0; nc < 8; ++nc) {
    const int n0 = nc * 128;
    __syncthreads();
    {  // transpose 128 (s) x 64 (d) K block into sKt[d][s]
      int sl = tid >> 1, db = (tid & 1) * 32;
      const unsigned short* src = &Ks[headBase + (size_t)(n0 + sl) * DK + db];
      union { uint4 v[4]; unsigned short hh[32]; } t;
      t.v[0] = *(const uint4*)(src + 0);
      t.v[1] = *(const uint4*)(src + 8);
      t.v[2] = *(const uint4*)(src + 16);
      t.v[3] = *(const uint4*)(src + 24);
#pragma unroll
      for (int e = 0; e < 32; ++e) sKt[(db + e) * SKT_STRIDE + sl] = t.hh[e];
    }
    __syncthreads();

    v8f c0 = vzero, c1 = vzero;
#pragma unroll
    for (int ks = 0; ks < 2; ++ks) {
      Frag a0, a1, bt;
      const unsigned short* pa0 = &sQ[al * SQ_STRIDE + ks * 32 + ahalf];
      a0.u4[0] = *(const uint4*)(pa0);
      a0.u4[1] = *(const uint4*)(pa0 + 16);
      const unsigned short* pa1 = &sQ[(16 + al) * SQ_STRIDE + ks * 32 + ahalf];
      a1.u4[0] = *(const uint4*)(pa1);
      a1.u4[1] = *(const uint4*)(pa1 + 16);
      const unsigned short* pb = &sKt[(ks * 32) * SKT_STRIDE + wave * 16 + al];
#pragma unroll
      for (int s = 0; s < 16; ++s) {
        int k = (s < 8) ? (ahalf + s) : (16 + ahalf + (s - 8));
        bt.bf[s] = __builtin_bit_cast(__bf16, pb[k * SKT_STRIDE]);
      }
      c0 = wmma_bf16(a0, bt, c0);
      c1 = wmma_bf16(a1, bt, c1);
    }
#pragma unroll
    for (int r = 0; r < 8; ++r) {
      sP[(r + 8 * hi) * SP_STRIDE + n0 + wave * 16 + al]      = c0[r];
      sP[(16 + r + 8 * hi) * SP_STRIDE + n0 + wave * 16 + al] = c1[r];
    }
  }
  __syncthreads();

  // ---- masked softmax; each wave owns 4 rows ----
  for (int i = 0; i < 4; ++i) {
    const int row = wave * 4 + i;
    const unsigned char* mrow = mask + ((size_t)b * S_LEN + q0 + row) * S_LEN;
    float vals[32];
    float mx = -3.0e38f;
#pragma unroll
    for (int j = 0; j < 32; ++j) {
      int c = lane + j * 32;
      float v = sP[row * SP_STRIDE + c];
      if (mrow[c]) v = -1.0e9f;
      vals[j] = v;
      mx = fmaxf(mx, v);
    }
#pragma unroll
    for (int o = 16; o > 0; o >>= 1) mx = fmaxf(mx, __shfl_xor(mx, o, 32));
    float sum = 0.f;
#pragma unroll
    for (int j = 0; j < 32; ++j) {
      float e = __expf(vals[j] - mx);
      vals[j] = e;
      sum += e;
    }
#pragma unroll
    for (int o = 16; o > 0; o >>= 1) sum += __shfl_xor(sum, o, 32);
    const float inv = 1.0f / sum;
    float* arow = attn_out + ((((size_t)(b * NH + h)) * S_LEN) + q0 + row) * S_LEN;
#pragma unroll
    for (int j = 0; j < 32; ++j) {
      int c = lane + j * 32;
      float p = vals[j] * inv;
      arow[c] = p;                              // single coalesced attn write
      sPu[row * (2 * SP_STRIDE) + c] = f2bf(p); // in-place bf16 repack (wave owns row)
    }
  }
  __syncthreads();

  // ---- ctx = P @ V : 8 tiles (2 M x 4 N16), one per wave ----
  const int wm = wave & 1, wn = wave >> 1;
  v8f cc = vzero;
  for (int kk = 0; kk < 1024; kk += 32) {
    Frag a, bt;
    const unsigned short* pa = &sPu[(wm * 16 + al) * (2 * SP_STRIDE) + kk + ahalf];
    a.u4[0] = *(const uint4*)(pa);
    a.u4[1] = *(const uint4*)(pa + 16);
    const unsigned short* pv = &Vs[headBase + (size_t)kk * DK + wn * 16 + al];
#pragma unroll
    for (int s = 0; s < 16; ++s) {
      int k = (s < 8) ? (ahalf + s) : (16 + ahalf + (s - 8));
      bt.bf[s] = __builtin_bit_cast(__bf16, pv[(size_t)k * DK]);
    }
    cc = wmma_bf16(a, bt, cc);
  }
#pragma unroll
  for (int r = 0; r < 8; ++r) {
    int m = q0 + wm * 16 + r + 8 * hi;
    int d = h * DK + wn * 16 + al;
    ctx[((size_t)b * S_LEN + m) * DM + d] = cc[r];
  }
}

// ---------------------------------------------------------------------------
// Kernel 3: out = LayerNorm(ctx + query). One block (256 thr) per row.
// ---------------------------------------------------------------------------
__global__ __launch_bounds__(256)
void ln_kernel(const float* __restrict__ query,
               const float* __restrict__ ctx,
               float* __restrict__ outp)
{
  __shared__ float red[8];
  const int tid = threadIdx.x, lane = tid & 31, wave = tid >> 5;
  const size_t row = blockIdx.x;

  const float4 q = *(const float4*)&query[row * DM + tid * 4];
  const float4 c = *(const float4*)&ctx[row * DM + tid * 4];
  float4 v;
  v.x = q.x + c.x; v.y = q.y + c.y; v.z = q.z + c.z; v.w = q.w + c.w;

  float s = v.x + v.y + v.z + v.w;
#pragma unroll
  for (int o = 16; o > 0; o >>= 1) s += __shfl_xor(s, o, 32);
  if (lane == 0) red[wave] = s;
  __syncthreads();
  float tot = 0.f;
#pragma unroll
  for (int i = 0; i < 8; ++i) tot += red[i];
  const float mean = tot * (1.0f / DM);

  const float dx = v.x - mean, dy = v.y - mean, dz = v.z - mean, dw = v.w - mean;
  float sq = dx * dx + dy * dy + dz * dz + dw * dw;
#pragma unroll
  for (int o = 16; o > 0; o >>= 1) sq += __shfl_xor(sq, o, 32);
  __syncthreads();
  if (lane == 0) red[wave] = sq;
  __syncthreads();
  float vtot = 0.f;
#pragma unroll
  for (int i = 0; i < 8; ++i) vtot += red[i];
  const float rstd = rsqrtf(vtot * (1.0f / DM) + 1e-5f);

  float4 o;
  o.x = dx * rstd; o.y = dy * rstd; o.z = dz * rstd; o.w = dw * rstd;
  *(float4*)&outp[row * DM + tid * 4] = o;
}

// ---------------------------------------------------------------------------
extern "C" void kernel_launch(void* const* d_in, const int* in_sizes, int n_in,
                              void* d_out, int out_size, void* d_ws, size_t ws_size,
                              hipStream_t stream)
{
  (void)in_sizes; (void)n_in; (void)out_size; (void)ws_size;

  const float* query   = (const float*)d_in[0];
  const float* context = (const float*)d_in[1];
  const unsigned char* mask = (const unsigned char*)d_in[2];
  const float* Wq = (const float*)d_in[3];
  const float* Wk = (const float*)d_in[4];
  const float* Wv = (const float*)d_in[5];

  float* out_normed = (float*)d_out;                                   // [B,S,D]
  float* out_attn   = out_normed + (size_t)BATCH * S_LEN * DM;         // [B,H,S,S]

  const size_t headElems = (size_t)BATCH * NH * S_LEN * DK;            // 4M bf16 each
  unsigned short* Qs = (unsigned short*)d_ws;
  unsigned short* Ks = Qs + headElems;
  unsigned short* Vs = Ks + headElems;
  float* ctx = (float*)(Vs + headElems);                               // [B,S,D] f32

  dim3 g1(DM / 64, (BATCH * S_LEN) / 128, 3);
  qkv_proj_kernel<<<g1, 256, 0, stream>>>(query, context, Wq, Wk, Wv, Qs, Ks, Vs);

  dim3 g2(S_LEN / 32, NH, BATCH);
  size_t lds2 = (size_t)32 * SP_STRIDE * 4 +
                (size_t)(32 * SQ_STRIDE + 64 * SKT_STRIDE) * 2;  // 153600 B
  attn_kernel<<<g2, 256, lds2, stream>>>(Qs, Ks, Vs, mask, out_attn, ctx);

  dim3 g3(BATCH * S_LEN);
  ln_kernel<<<g3, 256, 0, stream>>>(query, ctx, out_normed);
}
